// UniversalKANLayer_8117488189985
// MI455X (gfx1250) — compile-verified, matrix operand order checked by reading, not run on hardware
//
#include <hip/hip_runtime.h>
#include <hip/hip_bf16.h>

#define B_   256
#define OUT_ 256
#define IN_  256
#define G_   8
#define TABSTRIDE 18   // P[0..8], S[0..8]

typedef __attribute__((ext_vector_type(16))) __bf16        v16bf;
typedef __attribute__((ext_vector_type(8)))  float         v8f;
typedef __attribute__((ext_vector_type(4)))  unsigned int  u32x4;

__device__ __forceinline__ __bf16 to_bf16(float f) {
  unsigned u = __builtin_bit_cast(unsigned, f);
  unsigned r = u + 0x7FFFu + ((u >> 16) & 1u);      // round-to-nearest-even
  return __builtin_bit_cast(__bf16, (unsigned short)(r >> 16));
}

// ---------------------------------------------------------------------------
// Kernel 1: per-(o,i) prefix/suffix tables.
//   P[k] = sum_{g<k}  w[g] * exp(+s * grid[g])
//   S[k] = sum_{g>=k} w[g] * exp(-s * grid[g])
// so that  sum_g w[g]*exp(-s*|t-grid[g]|) = exp(-s*t)*P[k] + exp(s*t)*S[k]
// with k = #{g : grid[g] <= t}.
// ---------------------------------------------------------------------------
__global__ void kan_tables(const float* __restrict__ w,
                           const float* __restrict__ s,
                           const float* __restrict__ grid,
                           float* __restrict__ tab) {
  int idx = blockIdx.x * blockDim.x + threadIdx.x;   // (o*IN + i)
  if (idx >= OUT_ * IN_) return;
  float sc = s[idx];
  float wv[G_], gp[G_];
#pragma unroll
  for (int g = 0; g < G_; ++g) {
    wv[g] = w[idx * G_ + g];
    gp[g] = grid[g];
  }
  float P[G_ + 1], S[G_ + 1];
  P[0] = 0.f;
#pragma unroll
  for (int k = 1; k <= G_; ++k) P[k] = fmaf(wv[k - 1], __expf(sc * gp[k - 1]), P[k - 1]);
  S[G_] = 0.f;
#pragma unroll
  for (int k = G_ - 1; k >= 0; --k) S[k] = fmaf(wv[k], __expf(-sc * gp[k]), S[k + 1]);
  float* t = tab + (size_t)idx * TABSTRIDE;
#pragma unroll
  for (int k = 0; k <= G_; ++k) {
    t[k]     = P[k];
    t[9 + k] = S[k];
  }
}

// ---------------------------------------------------------------------------
// Kernel 2: xn = tanh(x) (f32 + bf16 copies), k = #{g : grid[g] <= xn}
// ---------------------------------------------------------------------------
__global__ void kan_prep(const float* __restrict__ x,
                         const float* __restrict__ grid,
                         float* __restrict__ xn,
                         int* __restrict__ kix,
                         __bf16* __restrict__ xnbf) {
  int idx = blockIdx.x * blockDim.x + threadIdx.x;   // (b*IN + i)
  if (idx >= B_ * IN_) return;
  float t = tanhf(x[idx]);
  int k = 0;
#pragma unroll
  for (int g = 0; g < G_; ++g) k += (grid[g] <= t) ? 1 : 0;
  xn[idx]   = t;
  kix[idx]  = k;
  xnbf[idx] = to_bf16(t);
}

// ---------------------------------------------------------------------------
// Kernel 3: base_activation -> bf16 (done once; reused by 16 WMMA tiles)
// ---------------------------------------------------------------------------
__global__ void kan_base_cvt(const float* __restrict__ basew,
                             __bf16* __restrict__ basebf) {
  int idx = blockIdx.x * blockDim.x + threadIdx.x;
  if (idx >= OUT_ * IN_) return;
  basebf[idx] = to_bf16(basew[idx]);
}

// ---------------------------------------------------------------------------
// Kernel 4: spline reduction. One thread per (b,o); 16b x 16o tile per block.
// Lanes 0..15 of a wave share (o,i) -> scaler/table loads are broadcasts;
// LDS rows padded to 65 so sx[lb*65+ii] hits distinct banks.
// 2 exps per (b,o,i) instead of the naive 8.
// ---------------------------------------------------------------------------
#define ITILE 64
__global__ void kan_main(const float* __restrict__ scaler,
                         const float* __restrict__ tab,
                         const float* __restrict__ xn,
                         const int* __restrict__ kix,
                         float* __restrict__ out) {
  __shared__ float sx[16 * (ITILE + 1)];
  __shared__ int   sk[16 * (ITILE + 1)];
  const int bt = (blockIdx.x & 15) * 16;
  const int ot = (blockIdx.x >> 4) * 16;
  const int lb = threadIdx.x & 15;
  const int lo = threadIdx.x >> 4;
  const int b = bt + lb;
  const int o = ot + lo;

  float acc = 0.f;
  for (int i0 = 0; i0 < IN_; i0 += ITILE) {
    for (int e = threadIdx.x; e < 16 * ITILE; e += 256) {
      int row = e >> 6, col = e & (ITILE - 1);
      sx[row * (ITILE + 1) + col] = xn[(bt + row) * IN_ + i0 + col];
      sk[row * (ITILE + 1) + col] = kix[(bt + row) * IN_ + i0 + col];
    }
    __syncthreads();
#pragma unroll 4
    for (int ii = 0; ii < ITILE; ++ii) {
      int   i  = i0 + ii;
      float t  = sx[lb * (ITILE + 1) + ii];
      int   k  = sk[lb * (ITILE + 1) + ii];
      float sc = scaler[o * IN_ + i];
      const float* tp = tab + (size_t)(o * IN_ + i) * TABSTRIDE;
      float P  = tp[k];
      float S  = tp[9 + k];
      float st = sc * t;
      float ep = __expf(st);
      float em = __expf(-st);
      acc = fmaf(em, P, acc);
      acc = fmaf(ep, S, acc);
    }
    __syncthreads();
  }
  out[b * OUT_ + o] = acc;
}

// ---------------------------------------------------------------------------
// Kernel 5: residual term  out[b,o] += sum_i base[o,i]*xn[b,i]  via WMMA bf16.
// One wave32 per 16x16 output tile. 16-bit A/B fragment layout: lane m=l&15,
// half=l>>4; VGPRs 0-3 hold K = half*8 .. half*8+7 (contiguous!), VGPRs 4-7
// hold K = 16+half*8 .. +7  ->  each fragment is two global_load_b128.
// ---------------------------------------------------------------------------
__device__ __forceinline__ v16bf load_frag(const __bf16* __restrict__ p,
                                           int off0, int off1) {
  union { v16bf v; u32x4 q[2]; } u;
  u.q[0] = *(const u32x4*)(p + off0);   // 8 bf16 = 16 bytes
  u.q[1] = *(const u32x4*)(p + off1);
  return u.v;
}

__global__ void kan_base_wmma(const __bf16* __restrict__ xnbf,
                              const __bf16* __restrict__ basebf,
                              float* __restrict__ out) {
  const int bt   = (blockIdx.x & 15) * 16;
  const int ot   = (blockIdx.x >> 4) * 16;
  const int l    = threadIdx.x;
  const int m    = l & 15;
  const int half = l >> 4;

  // C/D layout: VGPR r, lane l -> M = r + 8*half, N = m
  v8f c;
#pragma unroll
  for (int r = 0; r < 8; ++r)
    c[r] = out[(bt + r + 8 * half) * OUT_ + ot + m];

  const __bf16* xa = xnbf   + (bt + m) * IN_;   // A row (M = bt+m)
  const __bf16* bb = basebf + (ot + m) * IN_;   // B col (N = ot+m) = base row

#pragma unroll
  for (int kk = 0; kk < IN_ / 32; ++kk) {
    const int i0 = kk * 32;
    v16bf a  = load_frag(xa, i0 + half * 8, i0 + 16 + half * 8);
    v16bf bm = load_frag(bb, i0 + half * 8, i0 + 16 + half * 8);
    c = __builtin_amdgcn_wmma_f32_16x16x32_bf16(
        /*neg_a=*/false, a, /*neg_b=*/false, bm,
        /*c_mod=*/(short)0, c, /*reuse_a=*/false, /*reuse_b=*/false);
  }

#pragma unroll
  for (int r = 0; r < 8; ++r)
    out[(bt + r + 8 * half) * OUT_ + ot + m] = c[r];
}

// ---------------------------------------------------------------------------
extern "C" void kernel_launch(void* const* d_in, const int* in_sizes, int n_in,
                              void* d_out, int out_size, void* d_ws, size_t ws_size,
                              hipStream_t stream) {
  const float* x     = (const float*)d_in[0];   // (B, IN)
  const float* w     = (const float*)d_in[1];   // (OUT, IN, G)
  const float* s     = (const float*)d_in[2];   // (OUT, IN)
  const float* basew = (const float*)d_in[3];   // (OUT, IN)
  const float* grid  = (const float*)d_in[4];   // (G)
  float*       out   = (float*)d_out;           // (B, OUT)

  char* ws = (char*)d_ws;
  size_t off = 0;
  float* tab = (float*)(ws + off);  off += (size_t)OUT_ * IN_ * TABSTRIDE * 4;  // 4.5 MB
  float* xn  = (float*)(ws + off);  off += (size_t)B_ * IN_ * 4;
  int*   kix = (int*)  (ws + off);  off += (size_t)B_ * IN_ * 4;
  __bf16* xnbf   = (__bf16*)(ws + off); off += (size_t)B_ * IN_ * 2;
  __bf16* basebf = (__bf16*)(ws + off); off += (size_t)OUT_ * IN_ * 2;

  kan_tables  <<<(OUT_ * IN_) / 256, 256, 0, stream>>>(w, s, grid, tab);
  kan_prep    <<<(B_ * IN_) / 256, 256, 0, stream>>>(x, grid, xn, kix, xnbf);
  kan_base_cvt<<<(OUT_ * IN_) / 256, 256, 0, stream>>>(basew, basebf);
  kan_main    <<<(B_ / 16) * (OUT_ / 16), 256, 0, stream>>>(s, tab, xn, kix, out);
  kan_base_wmma<<<(B_ / 16) * (OUT_ / 16), 32, 0, stream>>>(xnbf, basebf, out);
}
